// Graph_Transformer_20959440404666
// MI455X (gfx1250) — matrix operand order repaired and checked by reading, not compiled, not run on hardware
//
#include <hip/hip_runtime.h>
#include <hip/hip_bf16.h>
#include <stdint.h>

// ---------------------------------------------------------------------------
// Graph Transformer for MI455X (gfx1250, wave32).
// Dense linear algebra via v_wmma_f32_16x16x32_f16 (f32 operands staged to LDS,
// converted to f16 at fragment load, f32 accumulate).
// Global->LDS tile movement uses the CDNA5 Tensor Data Mover with
// DOUBLE-BUFFERED software pipelining: the TDM for K-tile i+1 is issued before
// computing K-tile i, and tile completion is tracked with s_wait_tensorcnt
// relying on per-wave in-order TDM completion (2 tensor ops per tile).
// Falls back to vectorized float4 staging when the builtin is unavailable.
// All GEMM M sizes are multiples of 128 and all K/N sizes multiples of 4;
// the TDM path additionally gets OOB zero-fill for free from the descriptor's
// remaining-extent tensor dims.
// ---------------------------------------------------------------------------

typedef __attribute__((ext_vector_type(16))) _Float16 v16h;
typedef __attribute__((ext_vector_type(8)))  float    v8f;
typedef __attribute__((ext_vector_type(4)))  unsigned int u32x4;
typedef __attribute__((ext_vector_type(4)))  int      i32x4;
typedef __attribute__((ext_vector_type(8)))  int      i32x8;

#define TPB 256

#if defined(__gfx1250__) && defined(__has_builtin)
# if __has_builtin(__builtin_amdgcn_tensor_load_to_lds)
#  define GT_HAVE_TDM 1
# endif
#endif

#ifdef GT_HAVE_TDM
// Issue a 2D tile DMA: global (row-major, row pitch `stride_elems` f32) -> LDS
// (packed tile_w*4 bytes per row). `w_rem`/`h_rem` are the remaining tensor
// extents measured from the tile start, so OOB elements are zero-filled by TDM.
__device__ inline void gt_tdm_load_2d(const float* gsrc, unsigned lds_off,
                                      int w_rem, int h_rem, int tile_w,
                                      int tile_h, long long stride_elems) {
  unsigned long long ga = (unsigned long long)(uintptr_t)gsrc;
  unsigned long long s0 = (unsigned long long)stride_elems;
  unsigned td0 = (unsigned)w_rem, td1 = (unsigned)h_rem;
  u32x4 g0;
  g0[0] = 1u;                                           // count=1, user D#
  g0[1] = lds_off;                                      // lds_addr (bytes)
  g0[2] = (unsigned)(ga & 0xFFFFFFFFu);                 // global_addr[31:0]
  g0[3] = (unsigned)((ga >> 32) & 0x01FFFFFFu) | (2u << 30);  // [56:32] | type=2
  i32x8 g1;
  g1[0] = (int)(2u << 16);                              // data_size=4B; no pad
  g1[1] = (int)((td0 & 0xFFFFu) << 16);                 // tensor_dim0[15:0]
  g1[2] = (int)((td0 >> 16) | ((td1 & 0xFFFFu) << 16)); // td0 hi | td1 lo
  g1[3] = (int)((td1 >> 16) | ((unsigned)tile_w << 16));// td1 hi | tile_dim0
  g1[4] = (int)(unsigned)tile_h;                        // tile_dim1; tile_dim2=0
  g1[5] = (int)(s0 & 0xFFFFFFFFu);                      // dim0_stride[31:0]
  g1[6] = (int)((s0 >> 32) & 0xFFFFu);                  // stride hi | stride1 lo
  g1[7] = 0;
  i32x4 g2 = (i32x4)0, g3 = (i32x4)0;
#if __clang_major__ >= 23
  i32x8 gx = (i32x8)0;
  __builtin_amdgcn_tensor_load_to_lds(g0, g1, g2, g3, gx, 0);
#else
  __builtin_amdgcn_tensor_load_to_lds(g0, g1, g2, g3, 0);
#endif
}
#endif

// --------------------------- WMMA fragment loads ---------------------------
// A-matrix 16x32 f16 (ISA 7.12.2): lanes 0-15 row M=lane, K in {0..7,16..23};
// lanes 16-31 row M=lane-16, K in {8..15,24..31}; VGPR j holds pair (k,k+1).
// LDS holds fp32; convert while loading (packs to v_cvt_pk_f16_f32).
__device__ inline v16h gt_load_a32(const float* s, int ld, int mBase, int lane) {
  int row = mBase + (lane & 15);
  int ka  = (lane & 16) ? 8 : 0;
  v16h f;
#pragma unroll
  for (int j = 0; j < 8; ++j) {
    int k = ((j < 4) ? (2 * j) : (2 * j + 8)) + ka;
    f[2 * j]     = (_Float16)s[row * ld + k];
    f[2 * j + 1] = (_Float16)s[row * ld + k + 1];
  }
  return f;
}

// B-matrix 32x16 f16: lane = N; K order matches A: lanes 0-15 K in
// {0..7,16..23}, lanes 16-31 K in {8..15,24..31}; VGPR j holds pair (k,k+1).
__device__ inline v16h gt_load_b32(const float* s, int ld, int nBase, int lane) {
  int col = nBase + (lane & 15);
  int ka  = (lane & 16) ? 8 : 0;
  v16h f;
#pragma unroll
  for (int j = 0; j < 8; ++j) {
    int k = ((j < 4) ? (2 * j) : (2 * j + 8)) + ka;
    f[2 * j]     = (_Float16)s[k * ld + col];
    f[2 * j + 1] = (_Float16)s[(k + 1) * ld + col];
  }
  return f;
}

// ------------------------------- GEMM kernel -------------------------------
// C[M,N] = act( alpha * (A[M,K] @ W[K,N] + bias) ) + R   (bias/R optional)
// Requires: M % 128 == 0, K % 4 == 0, N % 4 == 0 (true for every call here).
__global__ __launch_bounds__(TPB) void gt_wmma_gemm(
    const float* __restrict__ A, const float* __restrict__ W,
    const float* __restrict__ bias, const float* __restrict__ R,
    float* __restrict__ C, int M, int N, int K, float alpha, int relu) {
  const int tid    = threadIdx.x;
  const int lane   = tid & 31;
  const int wave   = tid >> 5;
  const int blockM = blockIdx.y * 128;
  const int blockN = blockIdx.x * 64;
  const int wm     = wave >> 1;   // 0..3 -> 32-row slab
  const int wn     = wave & 1;    // 0..1 -> 32-col slab

  v8f acc[4];
#pragma unroll
  for (int f = 0; f < 4; ++f) { v8f z = {}; acc[f] = z; }

#ifdef GT_HAVE_TDM
  // Double-buffered, TDM-pipelined K loop: DMA of tile i+1 overlaps WMMA of i.
  __shared__ float sA[2][128 * 32];
  __shared__ float sB[2][32 * 64];
  const int ktiles = (K + 31) >> 5;

  if (wave == 0) {
    gt_tdm_load_2d(A + (size_t)blockM * K, (unsigned)(uintptr_t)&sA[0][0],
                   K, M - blockM, 32, 128, (long long)K);
    gt_tdm_load_2d(W + blockN, (unsigned)(uintptr_t)&sB[0][0],
                   N - blockN, K, 64, 32, (long long)N);
  }
  for (int i = 0; i < ktiles; ++i) {
    if (wave == 0) {
      if (i + 1 < ktiles) {
        int k1 = (i + 1) * 32;
        int nb = (i + 1) & 1;
        gt_tdm_load_2d(A + (size_t)blockM * K + k1,
                       (unsigned)(uintptr_t)&sA[nb][0],
                       K - k1, M - blockM, 32, 128, (long long)K);
        gt_tdm_load_2d(W + (size_t)k1 * N + blockN,
                       (unsigned)(uintptr_t)&sB[nb][0],
                       N - blockN, K - k1, 64, 32, (long long)N);
        // 2 TDM ops per tile, in-order completion per wave: <=2 outstanding
        // means tile i has fully landed in LDS.
        __builtin_amdgcn_s_wait_tensorcnt(2);
      } else {
        __builtin_amdgcn_s_wait_tensorcnt(0);
      }
    }
    __syncthreads();
    const float* cA = &sA[i & 1][0];
    const float* cB = &sB[i & 1][0];

    v16h a0 = gt_load_a32(cA, 32, wm * 32, lane);
    v16h a1 = gt_load_a32(cA, 32, wm * 32 + 16, lane);
    v16h b0 = gt_load_b32(cB, 64, wn * 32, lane);
    v16h b1 = gt_load_b32(cB, 64, wn * 32 + 16, lane);

    acc[0] = __builtin_amdgcn_wmma_f32_16x16x32_f16(false, a0, false, b0, (short)0, acc[0], false, false);
    acc[1] = __builtin_amdgcn_wmma_f32_16x16x32_f16(false, a0, false, b1, (short)0, acc[1], false, false);
    acc[2] = __builtin_amdgcn_wmma_f32_16x16x32_f16(false, a1, false, b0, (short)0, acc[2], false, false);
    acc[3] = __builtin_amdgcn_wmma_f32_16x16x32_f16(false, a1, false, b1, (short)0, acc[3], false, false);
    __syncthreads();  // allows wave0 to overwrite buffer (i+1)&1 next iter
  }
#else
  // Vectorized fallback staging: float4 granularity, quad-level guards.
  __shared__ float sA[128 * 32];
  __shared__ float sB[32 * 64];
  for (int k0 = 0; k0 < K; k0 += 32) {
    for (int q = tid; q < 128 * 32 / 4; q += TPB) {
      int idx = q * 4;
      int r = idx >> 5, c = idx & 31;
      int gr = blockM + r, gc = k0 + c;
      float4 v = make_float4(0.f, 0.f, 0.f, 0.f);
      if (gr < M && gc + 4 <= K) v = *(const float4*)(A + (size_t)gr * K + gc);
      *(float4*)(&sA[r * 32 + c]) = v;
    }
    for (int q = tid; q < 32 * 64 / 4; q += TPB) {
      int idx = q * 4;
      int r = idx >> 6, c = idx & 63;
      int gr = k0 + r, gc = blockN + c;
      float4 v = make_float4(0.f, 0.f, 0.f, 0.f);
      if (gr < K && gc + 4 <= N) v = *(const float4*)(W + (size_t)gr * N + gc);
      *(float4*)(&sB[r * 64 + c]) = v;
    }
    __syncthreads();

    v16h a0 = gt_load_a32(sA, 32, wm * 32, lane);
    v16h a1 = gt_load_a32(sA, 32, wm * 32 + 16, lane);
    v16h b0 = gt_load_b32(sB, 64, wn * 32, lane);
    v16h b1 = gt_load_b32(sB, 64, wn * 32 + 16, lane);

    acc[0] = __builtin_amdgcn_wmma_f32_16x16x32_f16(false, a0, false, b0, (short)0, acc[0], false, false);
    acc[1] = __builtin_amdgcn_wmma_f32_16x16x32_f16(false, a0, false, b1, (short)0, acc[1], false, false);
    acc[2] = __builtin_amdgcn_wmma_f32_16x16x32_f16(false, a1, false, b0, (short)0, acc[2], false, false);
    acc[3] = __builtin_amdgcn_wmma_f32_16x16x32_f16(false, a1, false, b1, (short)0, acc[3], false, false);
    __syncthreads();
  }
#endif

  // Epilogue. C/D layout: lane 0-15 N=lane, VGPR r -> M=r; lanes 16-31 M=r+8.
#pragma unroll
  for (int f = 0; f < 4; ++f) {
    int mB = blockM + wm * 32 + (f >> 1) * 16;
    int nB = blockN + wn * 32 + (f & 1) * 16;
    int n  = nB + (lane & 15);
    int m0 = mB + ((lane >> 4) << 3);
    v8f t = acc[f];
#pragma unroll
    for (int r = 0; r < 8; ++r) {
      int m = m0 + r;
      if (m < M && n < N) {
        float v = t[r];
        if (bias) v += bias[n];
        v *= alpha;
        if (relu) v = fmaxf(v, 0.0f);
        if (R) v += R[(size_t)m * N + n];
        C[(size_t)m * N + n] = v;
      }
    }
  }
}

// ------------------------------- LayerNorm ---------------------------------
__global__ __launch_bounds__(TPB) void gt_layernorm(
    const float* __restrict__ x, const float* __restrict__ g,
    const float* __restrict__ b, float* __restrict__ y, int d, int relu) {
  __shared__ float s1[TPB], s2[TPB];
  int row = blockIdx.x, tid = threadIdx.x;
  const float* xr = x + (size_t)row * d;
  float a = 0.f, q = 0.f;
  for (int j = tid; j < d; j += TPB) { float v = xr[j]; a += v; q += v * v; }
  s1[tid] = a; s2[tid] = q; __syncthreads();
  for (int s = TPB / 2; s > 0; s >>= 1) {
    if (tid < s) { s1[tid] += s1[tid + s]; s2[tid] += s2[tid + s]; }
    __syncthreads();
  }
  float mean = s1[0] / d;
  float var  = s2[0] / d - mean * mean;
  float rs   = rsqrtf(var + 1e-6f);
  for (int j = tid; j < d; j += TPB) {
    float v = g[j] * (xr[j] - mean) * rs + b[j];
    if (relu) v = fmaxf(v, 0.f);
    y[(size_t)row * d + j] = v;
  }
}

// LN + ReLU + typed masked accumulate: xt[row] += (types[row]==t) ? relu(ln(y)) : 0
__global__ __launch_bounds__(TPB) void gt_ln_relu_mask_add(
    const float* __restrict__ yin, const float* __restrict__ g,
    const float* __restrict__ b, const int* __restrict__ types, int t,
    float* __restrict__ xt, int d) {
  int row = blockIdx.x;
  if (types[row] != t) return;
  __shared__ float s1[TPB], s2[TPB];
  int tid = threadIdx.x;
  const float* xr = yin + (size_t)row * d;
  float a = 0.f, q = 0.f;
  for (int j = tid; j < d; j += TPB) { float v = xr[j]; a += v; q += v * v; }
  s1[tid] = a; s2[tid] = q; __syncthreads();
  for (int s = TPB / 2; s > 0; s >>= 1) {
    if (tid < s) { s1[tid] += s1[tid + s]; s2[tid] += s2[tid + s]; }
    __syncthreads();
  }
  float mean = s1[0] / d;
  float rs   = rsqrtf(s2[0] / d - mean * mean + 1e-6f);
  for (int j = tid; j < d; j += TPB) {
    float v = fmaxf(g[j] * (xr[j] - mean) * rs + b[j], 0.f);
    xt[(size_t)row * d + j] += v;
  }
}

// ----------------------- Multi-head attention (b,h) ------------------------
__global__ __launch_bounds__(TPB) void gt_mha(
    const float* __restrict__ q, const float* __restrict__ k,
    const float* __restrict__ v, const int* __restrict__ mask,
    float* __restrict__ o) {
  __shared__ float sq[64 * 32], sk[64 * 32], sv[64 * 32], sp[64 * 64];
  const int b = blockIdx.x >> 3, h = blockIdx.x & 7;
  const int tid = threadIdx.x;
  for (int idx = tid; idx < 64 * 32; idx += TPB) {
    int i = idx >> 5, d = idx & 31;
    size_t g = ((size_t)(b * 64 + i)) * 256 + h * 32 + d;
    sq[idx] = q[g]; sk[idx] = k[g]; sv[idx] = v[g];
  }
  __syncthreads();
  const float rsc = 0.17677669529663687f;  // 1/sqrt(32)
  for (int s = tid; s < 64 * 64; s += TPB) {
    int i = s >> 6, j = s & 63;
    float acc = 0.f;
#pragma unroll 8
    for (int d = 0; d < 32; ++d) acc += sq[i * 32 + d] * sk[j * 32 + d];
    sp[s] = (mask[b * 64 + j] == 0) ? -1e9f : acc * rsc;
  }
  __syncthreads();
  if (tid < 64) {
    float mx = -1e30f;
    for (int j = 0; j < 64; ++j) mx = fmaxf(mx, sp[tid * 64 + j]);
    float sum = 0.f;
    for (int j = 0; j < 64; ++j) { float e = expf(sp[tid * 64 + j] - mx); sp[tid * 64 + j] = e; sum += e; }
    float inv = 1.f / sum;
    for (int j = 0; j < 64; ++j) {
      float p = sp[tid * 64 + j] * inv;
      if (mask[b * 64 + j] == 0) p = 0.f;
      sp[tid * 64 + j] = p;
    }
  }
  __syncthreads();
  for (int idx = tid; idx < 64 * 32; idx += TPB) {
    int i = idx >> 5, d = idx & 31;
    float acc = 0.f;
#pragma unroll 8
    for (int j = 0; j < 64; ++j) acc += sp[i * 64 + j] * sv[j * 32 + d];
    o[((size_t)(b * 64 + i)) * 256 + h * 32 + d] = acc;
  }
}

// ------------------- Windowed edge message + scatter-add --------------------
// ax[e,d] = sum_k a[e,d,k] * xpad[e, d+k-32] (* ave[e,d]); msg[b,dst,d] += ax
__global__ __launch_bounds__(TPB) void gt_edge_msg(
    const float* __restrict__ a_chunk, const float* __restrict__ xn,
    const int* __restrict__ eidx, const float* __restrict__ ave,
    float* __restrict__ msg, int chunkStart, int totalE, int Eorig) {
  const int tid = threadIdx.x;
  const int r   = chunkStart + blockIdx.x;
  const int b   = r / totalE, e = r % totalE;
  int p0, p1;
  if (e < Eorig) { p0 = eidx[(b * Eorig + e) * 2 + 0]; p1 = eidx[(b * Eorig + e) * 2 + 1]; }
  else { int e2 = e - Eorig; p0 = eidx[(b * Eorig + e2) * 2 + 1]; p1 = eidx[(b * Eorig + e2) * 2 + 0]; }
  __shared__ float xrow[256];
  xrow[tid] = xn[((size_t)(b * 64 + p1)) * 256 + tid];
  __syncthreads();
  const float* arow = a_chunk + (size_t)blockIdx.x * 16384 + (size_t)tid * 64;
  float acc = 0.f;
#pragma unroll 8
  for (int kk = 0; kk < 64; ++kk) {
    int off = tid + kk - 32;
    float xv = (off >= 0 && off < 256) ? xrow[off] : 0.f;
    acc += arow[kk] * xv;
  }
  if (ave) acc *= ave[((size_t)(b * totalE + e)) * 256 + tid];
  atomicAdd(&msg[((size_t)(b * 64 + p0)) * 256 + tid], acc);
}

// --------------------------- angle MLP (1->64->256) -------------------------
__global__ __launch_bounds__(TPB) void gt_ang_mlp(
    const float* __restrict__ ang, const float* __restrict__ w1,
    const float* __restrict__ b1, const float* __restrict__ w2,
    const float* __restrict__ b2, float* __restrict__ out) {
  __shared__ float hid[64];
  const int row = blockIdx.x, tid = threadIdx.x;
  float a = ang[row];
  if (tid < 64) hid[tid] = fmaxf(w1[tid] * a + b1[tid], 0.f);
  __syncthreads();
  float acc = b2[tid];
#pragma unroll 8
  for (int h = 0; h < 64; ++h) acc += hid[h] * w2[h * 256 + tid];
  out[(size_t)row * 256 + tid] = acc;
}

// --------------------------- scatter-mean helpers ---------------------------
__global__ __launch_bounds__(TPB) void gt_scat_add(
    const float* __restrict__ attn, const int* __restrict__ aidx,
    float* __restrict__ sums, float* __restrict__ cnt) {
  const int row = blockIdx.x, tid = threadIdx.x;
  const int b = row / 256;
  const int seg = aidx[row];
  atomicAdd(&sums[((size_t)(b * 192 + seg)) * 256 + tid], attn[(size_t)row * 256 + tid]);
  if (tid == 0) atomicAdd(&cnt[b * 192 + seg], 1.0f);
}

__global__ __launch_bounds__(TPB) void gt_ave_fin(
    const float* __restrict__ sums, const float* __restrict__ cnt,
    float* __restrict__ ave) {
  const int row = blockIdx.x, tid = threadIdx.x;
  float c = cnt[row];
  ave[(size_t)row * 256 + tid] = (c > 0.f) ? sums[(size_t)row * 256 + tid] / c : 1.0f;
}

// ------------------------------ small utility -------------------------------
__global__ void gt_zero(float* p, int n) {
  int i = blockIdx.x * TPB + threadIdx.x;
  if (i < n) p[i] = 0.f;
}
__global__ void gt_add(float* dst, const float* src, int n) {
  int i = blockIdx.x * TPB + threadIdx.x;
  if (i < n) dst[i] += src[i];
}
__global__ void gt_dup_hid(const float* __restrict__ hid, float* __restrict__ dup,
                           int E, int total) {
  int i = blockIdx.x * TPB + threadIdx.x;
  if (i >= total) return;
  int c = i & 127;
  int r = i >> 7;
  int e = r % (2 * E);
  int b = r / (2 * E);
  dup[i] = hid[((size_t)(b * E + (e % E))) * 128 + c];
}
__global__ __launch_bounds__(TPB) void gt_gather_feat(
    const float* __restrict__ x, const int* __restrict__ sc_idx,
    const float* __restrict__ sc_mol, float* __restrict__ feat) {
  const int row = blockIdx.x, tid = threadIdx.x;
  const int b = row / 64, s = row % 64;
  const int i0 = sc_idx[(b * 64 + s) * 2 + 0];
  const int i1 = sc_idx[(b * 64 + s) * 2 + 1];
  for (int d = tid; d < 516; d += TPB) {
    float v;
    if (d < 256)      v = x[((size_t)(b * 64 + i0)) * 256 + d];
    else if (d < 512) v = x[((size_t)(b * 64 + i1)) * 256 + (d - 256)];
    else              v = sc_mol[(b * 64 + s) * 4 + (d - 512)];
    feat[(size_t)row * 516 + d] = v;
  }
}
__global__ void gt_final_dot(const float* __restrict__ z, const float* __restrict__ w,
                             const float* __restrict__ bptr, float* __restrict__ out,
                             int rows) {
  int r = blockIdx.x * TPB + threadIdx.x;
  if (r >= rows) return;
  float s = bptr[0];
#pragma unroll 8
  for (int kk = 0; kk < 128; ++kk) s += z[(size_t)r * 128 + kk] * w[kk];
  out[r] = s;
}

// ------------------------------- host driver --------------------------------
// Param pytree flattened in JAX canonical order (dict keys sorted, lists in
// order, {'b','w'} -> b first, {'b','g'} -> b first):
//   attn[t] (t=0..2, base=16t): ff0.b,ff0.w,ff1.b,ff1.w, l0.b,l0.w..l3.b,l3.w,
//                               ln0.b,ln0.g,ln1.b,ln1.g
//   48: bond_ann (b1,w1,b2,w2)   52: bond_enn (b1,w1,b2,w2)   56: enc_ln (b,g)
//   58: head: c1.b,c1.w, c1_ln.b,c1_ln.g, c2.b,c2.w, pre.b,pre.w,
//             pre_ln.b,pre_ln.g, types[i] (68+4i): lin.b,lin.w,ln.b,ln.g
//   100: mp_lin[j] (b,w)x6   112: mp_ln[j] (b,g)x6   124: proj (b,w)
//   126: sc_enn (b1,w1,b2,w2)
//   130..139: atom_x, bond_x, sc_pair_x, sc_mol_x, angles, mask, bond_idx,
//             sc_idx, angles_idx, sc_types
extern "C" void kernel_launch(void* const* d_in, const int* in_sizes, int n_in,
                              void* d_out, int out_size, void* d_ws, size_t ws_size,
                              hipStream_t stream) {
  (void)in_sizes; (void)n_in; (void)out_size; (void)ws_size;
  auto PF = [&](int i) -> const float* { return (const float*)d_in[i]; };

  const float* atom_x    = PF(130);
  const float* bond_x    = PF(131);
  const float* sc_pair_x = PF(132);
  const float* sc_mol_x  = PF(133);
  const float* angles    = PF(134);
  const int*   maskp     = (const int*)d_in[135];
  const int*   bond_idx  = (const int*)d_in[136];
  const int*   sc_idx    = (const int*)d_in[137];
  const int*   ang_idx   = (const int*)d_in[138];
  const int*   sc_types  = (const int*)d_in[139];

  float* w = (float*)d_ws;
  size_t off = 0;
  auto alloc = [&](size_t n) { float* p = w + off; off += n; return p; };
  float* x    = alloc(262144);        // 16*64*256
  float* xn   = alloc(262144);
  float* msg  = alloc(262144);
  float* qb   = alloc(262144);
  float* kb   = alloc(262144);
  float* vb   = alloc(262144);
  float* ob   = alloc(262144);
  float* ffh  = alloc(1048576);       // 16*64*1024
  float* hidB = alloc(1536 * 128);
  float* dupB = alloc(3072 * 128);
  float* hidS = alloc(1024 * 128);
  float* dupS = alloc(2048 * 128);
  float* achk = alloc(256 * 16384);   // 16 MB chunk of a_mat (L2-resident)
  float* angA = alloc(4096 * 256);
  float* aaS  = alloc(16 * 192 * 256);
  float* aaC  = alloc(16 * 192);
  float* aveA = alloc(16 * 192 * 256);
  float* feat = alloc(1024 * 516);
  float* xpre = alloc(1024 * 1024);
  float* xus  = alloc(1024 * 1024);
  float* xt   = alloc(1024 * 516);
  float* ybuf = alloc(1024 * 516);
  float* zc1  = alloc(1024 * 128);
  float* zc1b = alloc(1024 * 128);

  auto gemm = [&](const float* A, const float* W, const float* bias,
                  const float* R, float* C, int M, int N, int K,
                  float alpha, int relu) {
    dim3 g((N + 63) / 64, (M + 127) / 128);
    gt_wmma_gemm<<<g, TPB, 0, stream>>>(A, W, bias, R, C, M, N, K, alpha, relu);
  };
  auto zero = [&](float* p, int n) {
    gt_zero<<<(n + TPB - 1) / TPB, TPB, 0, stream>>>(p, n);
  };

  // ---- angle attention MLP + scatter-mean ----
  gt_ang_mlp<<<4096, TPB, 0, stream>>>(angles, PF(49), PF(48), PF(51), PF(50), angA);
  zero(aaS, 16 * 192 * 256);
  zero(aaC, 16 * 192);
  gt_scat_add<<<4096, TPB, 0, stream>>>(angA, ang_idx, aaS, aaC);
  gt_ave_fin<<<16 * 192, TPB, 0, stream>>>(aaS, aaC, aveA);

  // ---- edge-network hidden layers (relu(ex @ W1 + b1)) ----
  gemm(bond_x, PF(53), PF(52), nullptr, hidB, 1536, 128, 16, 1.f, 1);
  gemm(sc_pair_x, PF(127), PF(126), nullptr, hidS, 1024, 128, 8, 1.f, 1);
  gt_dup_hid<<<(3072 * 128 + TPB - 1) / TPB, TPB, 0, stream>>>(hidB, dupB, 96, 3072 * 128);
  gt_dup_hid<<<(2048 * 128 + TPB - 1) / TPB, TPB, 0, stream>>>(hidS, dupS, 64, 2048 * 128);

  // ---- input projection ----
  gemm(atom_x, PF(125), PF(124), nullptr, x, 1024, 256, 64, 1.f, 0);

  const float inv_sqrt_ksz = 0.125f;  // 1/sqrt(64)
  for (int t = 0; t < 3; ++t) {
    const int ab = 16 * t;
    // --- bond message pass ---
    gt_layernorm<<<1024, TPB, 0, stream>>>(x, PF(113 + 4 * t), PF(112 + 4 * t), xn, 256, 0);
    zero(msg, 262144);
    for (int c = 0; c < 12; ++c) {   // 3072 rows / 256
      gemm(dupB + (size_t)c * 256 * 128, PF(55), PF(54), nullptr, achk,
           256, 16384, 128, inv_sqrt_ksz, 0);
      gt_edge_msg<<<256, TPB, 0, stream>>>(achk, xn, bond_idx, aveA, msg, c * 256, 192, 96);
    }
    gemm(msg, PF(101 + 4 * t), PF(100 + 4 * t), x, x, 1024, 256, 256, 1.f, 0);

    // --- sc message pass ---
    gt_layernorm<<<1024, TPB, 0, stream>>>(x, PF(115 + 4 * t), PF(114 + 4 * t), xn, 256, 0);
    zero(msg, 262144);
    for (int c = 0; c < 8; ++c) {    // 2048 rows / 256
      gemm(dupS + (size_t)c * 256 * 128, PF(129), PF(128), nullptr, achk,
           256, 16384, 128, inv_sqrt_ksz, 0);
      gt_edge_msg<<<256, TPB, 0, stream>>>(achk, xn, sc_idx, nullptr, msg, c * 256, 128, 64);
    }
    gemm(msg, PF(103 + 4 * t), PF(102 + 4 * t), x, x, 1024, 256, 256, 1.f, 0);

    // --- attention block ---
    gt_layernorm<<<1024, TPB, 0, stream>>>(x, PF(ab + 13), PF(ab + 12), xn, 256, 0);
    gemm(xn, PF(ab + 5), PF(ab + 4), nullptr, qb, 1024, 256, 256, 1.f, 0);
    gemm(xn, PF(ab + 7), PF(ab + 6), nullptr, kb, 1024, 256, 256, 1.f, 0);
    gemm(xn, PF(ab + 9), PF(ab + 8), nullptr, vb, 1024, 256, 256, 1.f, 0);
    gt_mha<<<16 * 8, TPB, 0, stream>>>(qb, kb, vb, maskp, ob);
    gemm(ob, PF(ab + 11), PF(ab + 10), x, x, 1024, 256, 256, 1.f, 0);

    // --- feed-forward block ---
    gt_layernorm<<<1024, TPB, 0, stream>>>(x, PF(ab + 15), PF(ab + 14), xn, 256, 0);
    gemm(xn, PF(ab + 1), PF(ab + 0), nullptr, ffh, 1024, 1024, 256, 1.f, 1);
    gemm(ffh, PF(ab + 3), PF(ab + 2), x, x, 1024, 256, 1024, 1.f, 0);
  }

  // ---- head ----
  gt_layernorm<<<1024, TPB, 0, stream>>>(x, PF(57), PF(56), xn, 256, 0);
  gt_gather_feat<<<1024, TPB, 0, stream>>>(xn, sc_idx, sc_mol_x, feat);
  gemm(feat, PF(65), PF(64), nullptr, xpre, 1024, 1024, 516, 1.f, 0);
  gt_layernorm<<<1024, TPB, 0, stream>>>(xpre, PF(67), PF(66), xus, 1024, 1);
  zero(xt, 1024 * 516);
  for (int i = 0; i < 8; ++i) {
    const int tb = 68 + i * 4;
    gemm(xus, PF(tb + 1), PF(tb + 0), nullptr, ybuf, 1024, 516, 1024, 1.f, 0);
    gt_ln_relu_mask_add<<<1024, TPB, 0, stream>>>(ybuf, PF(tb + 3), PF(tb + 2),
                                                  sc_types, i, xt, 516);
  }
  gt_add<<<(1024 * 516 + TPB - 1) / TPB, TPB, 0, stream>>>(xt, feat, 1024 * 516);
  gemm(xt, PF(59), PF(58), nullptr, zc1, 1024, 128, 516, 1.f, 0);
  gt_layernorm<<<1024, TPB, 0, stream>>>(zc1, PF(61), PF(60), zc1b, 128, 1);
  gt_final_dot<<<4, TPB, 0, stream>>>(zc1b, PF(63), PF(62), (float*)d_out, 1024);
}
